// MultiHeadLatentAttention_10900626997357
// MI455X (gfx1250) — matrix operand order repaired and checked by reading, not compiled, not run on hardware
//
#include <hip/hip_runtime.h>
#include <hip/hip_bf16.h>
#include <stdint.h>

// ---------------------------------------------------------------------------
// Types for CDNA5 WMMA (wave32): A/B = 16 bf16 per lane, C/D = 8 f32 per lane
// ---------------------------------------------------------------------------
typedef __bf16 v16bf __attribute__((ext_vector_type(16)));
typedef float  v8f   __attribute__((ext_vector_type(8)));

union FragU { uint4 u[2]; v16bf bf; };

__device__ __forceinline__ unsigned short f2bf(float f) {
    union { float f; unsigned int u; } v; v.f = f;
    unsigned int r = v.u + 0x7FFFu + ((v.u >> 16) & 1u);  // round-nearest-even
    return (unsigned short)(r >> 16);
}

// gfx1250 async global->LDS copy, 16B per lane, tracked by ASYNCcnt.
// lds_off = byte offset within workgroup LDS (flat-pointer low 32 bits).
__device__ __forceinline__ void async_ld_b128(unsigned lds_off, const void* gaddr) {
    asm volatile("global_load_async_to_lds_b128 %0, %1, off"
                 :: "v"(lds_off), "v"((unsigned long long)(uintptr_t)gaddr)
                 : "memory");
}
__device__ __forceinline__ unsigned lds_off(const void* p) {
    return (unsigned)(uintptr_t)p;   // flat LDS address truncates to AS3 offset
}
__device__ __forceinline__ void wait_async0() {
    asm volatile("s_wait_asynccnt 0" ::: "memory");
}

// A-fragment: 16x32 bf16 tile, LDS layout [m][k], pitch multiple of 8 halfwords.
// ISA layout: lanes 0-15 m=0..15, VGPR0-3 k=(lh*8)+0..7, VGPR4-7 k=16+(lh*8)+0..7
// -> two contiguous 16B reads per lane.
__device__ __forceinline__ v16bf load_a_frag(const unsigned short* smem, int pitch) {
    int lane = threadIdx.x & 31;
    int m = lane & 15, lh = lane >> 4;
    const unsigned short* p = smem + m * pitch + lh * 8;
    FragU f;
    f.u[0] = *(const uint4*)(p);
    f.u[1] = *(const uint4*)(p + 16);
    return f.bf;
}

// B-fragment: 32x16 bf16 tile, LDS layout transposed [n][k], pitch mult of 8.
// lane: n = (lane&15)+n0, k = (lane>>4)*8 + e (e<8), 16 + (lane>>4)*8 + (e-8) (e>=8).
__device__ __forceinline__ v16bf load_b_frag(const unsigned short* smem, int pitch, int n0) {
    int lane = threadIdx.x & 31;
    int n = n0 + (lane & 15), kh = (lane >> 4) * 8;
    const unsigned short* p = smem + n * pitch + kh;
    FragU f;
    f.u[0] = *(const uint4*)(p);
    f.u[1] = *(const uint4*)(p + 16);
    return f.bf;
}

#define WMMA_BF16(A, B, C) \
    __builtin_amdgcn_wmma_f32_16x16x32_bf16(false, (A), false, (B), (short)0, (C), false, false)

// ---------------------------------------------------------------------------
// Elementwise f32 -> bf16 convert
// ---------------------------------------------------------------------------
__global__ void cvt_f32_bf16(const float* __restrict__ X, unsigned short* __restrict__ Y, int n) {
    int i = blockIdx.x * blockDim.x + threadIdx.x;
    if (i < n) Y[i] = f2bf(X[i]);
}

// ---------------------------------------------------------------------------
// RoPE (interleaved rotate-half, HEAD_DIM=128, theta=10000) + f32->bf16
// ---------------------------------------------------------------------------
__global__ void rope_to_bf16(const float* __restrict__ X, unsigned short* __restrict__ Y, int npairs) {
    int i = blockIdx.x * blockDim.x + threadIdx.x;
    if (i >= npairs) return;
    int row = i >> 10;                 // 1024 pairs per 2048-wide row
    int col = (i & 1023) * 2;
    float t = (float)(row & 2047);     // sequence position
    int d0 = col & 127, d1 = d0 + 1;
    const float LTH = 0.14391157f;     // ln(10000)/64
    float a0 = t * __expf(-(float)(d0 & 63) * LTH);
    float a1 = t * __expf(-(float)(d1 & 63) * LTH);
    float s0, c0, s1, c1;
    __sincosf(a0, &s0, &c0);
    __sincosf(a1, &s1, &c1);
    float x0 = X[row * 2048 + col];
    float x1 = X[row * 2048 + col + 1];
    Y[row * 2048 + col]     = f2bf(x0 * c0 - x1 * s0);
    Y[row * 2048 + col + 1] = f2bf(x1 * c1 + x0 * s1);
}

// ---------------------------------------------------------------------------
// bf16 GEMM: C(f32, MxN) = A(bf16, MxK, row-major) * B(bf16, KxN, row-major)
// Block tile 128x128x32, 256 threads = 8 waves; wave tile 32x64 (2x4 WMMA).
// 2-stage pipeline: A tile via async global->LDS, B tile global->reg (issued
// before compute) -> transposed LDS store after compute; one barrier/iter.
// ---------------------------------------------------------------------------
static constexpr int GBM = 128, GBN = 128, GBK = 32;
static constexpr int GAP = 40;   // A LDS pitch (halfwords) — 80B rows, 16B aligned
static constexpr int GBP = 40;   // B^T LDS pitch

__global__ __launch_bounds__(256)
void gemm_bf16(const unsigned short* __restrict__ A,
               const unsigned short* __restrict__ B,
               float* __restrict__ C,
               int M, int N, int K, int lda, int ldb, int ldc) {
    __shared__ unsigned short sA[2][GBM * GAP];   // [m][k], double-buffered
    __shared__ unsigned short sB[2][GBN * GBP];   // [n][k] (transposed)

    int tid = threadIdx.x;
    int lane = tid & 31, wave = tid >> 5;
    int wm = wave & 3, wn = wave >> 2;            // 4 waves along M, 2 along N
    int bn0 = blockIdx.x * GBN, bm0 = blockIdx.y * GBM;

    v8f zero8 = {0.f, 0.f, 0.f, 0.f, 0.f, 0.f, 0.f, 0.f};
    v8f acc[2][4];
#pragma unroll
    for (int i = 0; i < 2; ++i)
#pragma unroll
        for (int j = 0; j < 4; ++j) acc[i][j] = zero8;

    int arow = tid >> 2, acol = (tid & 3) * 8;    // A tile: 128 x 32
    int brow = tid >> 4, bcol = (tid & 15) * 8;   // B tile: 32 x 128
    const unsigned short* Ap0 = A + (bm0 + arow) * lda + acol;
    const unsigned short* Ap1 = A + (bm0 + 64 + arow) * lda + acol;
    const unsigned short* Bp0 = B + brow * ldb + bn0 + bcol;
    const unsigned short* Bp1 = B + (16 + brow) * ldb + bn0 + bcol;

    // ---- prologue: stage tile 0 into buffer 0 ----
    async_ld_b128(lds_off(&sA[0][arow * GAP + acol]),        Ap0);
    async_ld_b128(lds_off(&sA[0][(64 + arow) * GAP + acol]), Ap1);
    {
        unsigned short t[8];
        uint4 b0 = *(const uint4*)Bp0;
        uint4 b1 = *(const uint4*)Bp1;
        *(uint4*)t = b0;
#pragma unroll
        for (int j = 0; j < 8; ++j) sB[0][(bcol + j) * GBP + brow] = t[j];
        *(uint4*)t = b1;
#pragma unroll
        for (int j = 0; j < 8; ++j) sB[0][(bcol + j) * GBP + 16 + brow] = t[j];
    }
    wait_async0();
    __syncthreads();

    int Kt = K / GBK;
    for (int kt = 0; kt < Kt; ++kt) {
        int cur = kt & 1, nxt = cur ^ 1;
        bool more = (kt + 1) < Kt;
        uint4 b0 = {}, b1 = {};
        if (more) {
            int k1 = (kt + 1) * GBK;
            // next A tile: fire-and-forget async copies into the other buffer
            async_ld_b128(lds_off(&sA[nxt][arow * GAP + acol]),        Ap0 + k1);
            async_ld_b128(lds_off(&sA[nxt][(64 + arow) * GAP + acol]), Ap1 + k1);
            // next B tile: global loads in flight across the WMMA stage
            b0 = *(const uint4*)(Bp0 + (size_t)k1 * ldb);
            b1 = *(const uint4*)(Bp1 + (size_t)k1 * ldb);
            if (kt + 2 < Kt)
                __builtin_prefetch(Bp0 + (size_t)(kt + 2) * GBK * ldb, 0, 3);
        }

        // ---- compute on current buffer ----
        v16bf af0 = load_a_frag(sA[cur] + (wm * 32) * GAP, GAP);
        v16bf af1 = load_a_frag(sA[cur] + (wm * 32 + 16) * GAP, GAP);
#pragma unroll
        for (int j = 0; j < 4; ++j) {
            v16bf bfj = load_b_frag(sB[cur], GBP, wn * 64 + j * 16);
            acc[0][j] = WMMA_BF16(af0, bfj, acc[0][j]);
            acc[1][j] = WMMA_BF16(af1, bfj, acc[1][j]);
        }

        if (more) {   // B transpose store after compute (LDS-fast)
            unsigned short t[8];
            *(uint4*)t = b0;
#pragma unroll
            for (int j = 0; j < 8; ++j) sB[nxt][(bcol + j) * GBP + brow] = t[j];
            *(uint4*)t = b1;
#pragma unroll
            for (int j = 0; j < 8; ++j) sB[nxt][(bcol + j) * GBP + 16 + brow] = t[j];
        }
        wait_async0();     // next A tile resident
        __syncthreads();   // everyone done reading cur / writing nxt
    }

    // Store: D layout -> lane n = lane&15; VGPR r -> m = r + 8*(lane>>4)
    int lh = lane >> 4, nl = lane & 15;
#pragma unroll
    for (int i = 0; i < 2; ++i)
#pragma unroll
        for (int j = 0; j < 4; ++j) {
            float* Cp = C + (bm0 + wm * 32 + i * 16 + lh * 8) * ldc + bn0 + wn * 64 + j * 16 + nl;
#pragma unroll
            for (int r = 0; r < 8; ++r) Cp[r * ldc] = acc[i][j][r];
        }
}

// ---------------------------------------------------------------------------
// Fused flash attention: one workgroup = (b,h, 128 q-rows); 8 waves x 16 rows.
// Q and K tiles staged with async global->LDS; V transposed through VGPRs.
// ---------------------------------------------------------------------------
static constexpr int FQ = 128, FKV = 32;
static constexpr int QP = 136;   // 272B rows (16B-multiple)
static constexpr int KP = 136;
static constexpr int VP = 40;    // V^T pitch
static constexpr int PP = 40;    // P pitch

__global__ __launch_bounds__(256)
void mla_flash(const unsigned short* __restrict__ Q,
               const unsigned short* __restrict__ Kmat,
               const unsigned short* __restrict__ V,
               unsigned short* __restrict__ O) {
    __shared__ unsigned short sQ[FQ * QP];        // [q][d]   34816 B
    __shared__ unsigned short sK[FKV * KP];       // [kk][d]   8704 B (B-frag layout for QK^T)
    __shared__ unsigned short sV[128 * VP];       // [d][kk]  10240 B (B-frag layout for PV)
    __shared__ unsigned short sP[8 * 16 * PP];    // per-wave 16x32 P patch, 10240 B

    int tid = threadIdx.x, lane = tid & 31, wave = tid >> 5;
    int lh = lane >> 4, nl = lane & 15;
    int q0 = blockIdx.x * FQ;
    int b = blockIdx.y >> 4, h = blockIdx.y & 15;
    int rowbase = b * 2048;

    const unsigned short* Qg = Q + (rowbase + q0) * 2048 + h * 128;
    // Q tile 128x128 via async copies (8 per thread)
    for (int u = tid; u < 2048; u += 256) {
        int r = u >> 4, c = (u & 15) * 8;
        async_ld_b128(lds_off(&sQ[r * QP + c]), Qg + r * 2048 + c);
    }
    wait_async0();
    __syncthreads();

    // Hoist this wave's Q fragments (16 rows x full 128 d)
    v16bf qf[4];
#pragma unroll
    for (int dk = 0; dk < 4; ++dk)
        qf[dk] = load_a_frag(sQ + (wave * 16) * QP + dk * 32, QP);

    v8f zero8 = {0.f, 0.f, 0.f, 0.f, 0.f, 0.f, 0.f, 0.f};
    v8f acco[8];
#pragma unroll
    for (int jo = 0; jo < 8; ++jo) acco[jo] = zero8;
    float mrow[8], lrow[8];
#pragma unroll
    for (int r = 0; r < 8; ++r) { mrow[r] = -3.0e38f; lrow[r] = 0.f; }

    const float sm_scale = 0.08838834764831845f;   // 1/sqrt(128)

    for (int kv0 = 0; kv0 < 2048; kv0 += FKV) {
        const unsigned short* Kg = Kmat + (rowbase + kv0) * 2048 + h * 128;
        const unsigned short* Vg = V    + (rowbase + kv0) * 2048 + h * 128;
        // K chunk 32x128 row-major: async global->LDS (2 per thread)
        {
            int r = tid >> 4, c = (tid & 15) * 8;
            async_ld_b128(lds_off(&sK[r * KP + c]), Kg + r * 2048 + c);
            int u = tid + 256; int r2 = u >> 4, c2 = (u & 15) * 8;
            async_ld_b128(lds_off(&sK[r2 * KP + c2]), Kg + r2 * 2048 + c2);
        }
        // V chunk transposed into [d][kk] (through VGPRs)
        {
            unsigned short t[8];
            int u = tid;          int r = u >> 4, c = (u & 15) * 8;
            *(uint4*)t = *(const uint4*)(Vg + r * 2048 + c);
#pragma unroll
            for (int j = 0; j < 8; ++j) sV[(c + j) * VP + r] = t[j];
            u = tid + 256;        r = u >> 4;  c = (u & 15) * 8;
            *(uint4*)t = *(const uint4*)(Vg + r * 2048 + c);
#pragma unroll
            for (int j = 0; j < 8; ++j) sV[(c + j) * VP + r] = t[j];
        }
        wait_async0();
        __syncthreads();

        // S strip: 16 rows x 32 keys
        v8f s0 = zero8, s1 = zero8;
#pragma unroll
        for (int dk = 0; dk < 4; ++dk) {
            v16bf bk0 = load_b_frag(sK + dk * 32, KP, 0);
            v16bf bk1 = load_b_frag(sK + dk * 32, KP, 16);
            s0 = WMMA_BF16(qf[dk], bk0, s0);
            s1 = WMMA_BF16(qf[dk], bk1, s1);
        }

        // Online softmax (a row's 16 values live across one 16-lane half-group)
        unsigned short* pw = sP + wave * 16 * PP;
#pragma unroll
        for (int r = 0; r < 8; ++r) {
            float v0 = s0[r] * sm_scale, v1 = s1[r] * sm_scale;
            float mx = fmaxf(v0, v1);
            mx = fmaxf(mx, __shfl_xor(mx, 1, 32));
            mx = fmaxf(mx, __shfl_xor(mx, 2, 32));
            mx = fmaxf(mx, __shfl_xor(mx, 4, 32));
            mx = fmaxf(mx, __shfl_xor(mx, 8, 32));
            float mnew = fmaxf(mrow[r], mx);
            float corr = __expf(mrow[r] - mnew);
            float p0 = __expf(v0 - mnew), p1 = __expf(v1 - mnew);
            float ps = p0 + p1;
            ps += __shfl_xor(ps, 1, 32);
            ps += __shfl_xor(ps, 2, 32);
            ps += __shfl_xor(ps, 4, 32);
            ps += __shfl_xor(ps, 8, 32);
            lrow[r] = lrow[r] * corr + ps;
            mrow[r] = mnew;
#pragma unroll
            for (int jo = 0; jo < 8; ++jo) acco[jo][r] *= corr;
            int m = r + lh * 8;
            pw[m * PP + nl]      = f2bf(p0);
            pw[m * PP + 16 + nl] = f2bf(p1);
        }
        // wave-private LDS RAW: P stores must land before the frag reads
        asm volatile("s_wait_dscnt 0" ::: "memory");

        v16bf pf = load_a_frag(pw, PP);
#pragma unroll
        for (int jo = 0; jo < 8; ++jo) {
            v16bf vfj = load_b_frag(sV, VP, jo * 16);
            acco[jo] = WMMA_BF16(pf, vfj, acco[jo]);
        }
        __syncthreads();   // before next chunk overwrites sK/sV
    }

    // Normalize and emit bf16 ctx at (b, s, h*128+d)
    unsigned short* Og = O + (rowbase + q0 + wave * 16) * 2048 + h * 128;
#pragma unroll
    for (int r = 0; r < 8; ++r) {
        float inv = 1.0f / lrow[r];
        int m = r + lh * 8;
#pragma unroll
        for (int jo = 0; jo < 8; ++jo)
            Og[m * 2048 + jo * 16 + nl] = f2bf(acco[jo][r] * inv);
    }
}

// ---------------------------------------------------------------------------
// Host orchestration
// ---------------------------------------------------------------------------
extern "C" void kernel_launch(void* const* d_in, const int* in_sizes, int n_in,
                              void* d_out, int out_size, void* d_ws, size_t ws_size,
                              hipStream_t stream) {
    (void)in_sizes; (void)n_in; (void)out_size; (void)ws_size;
    const float* x    = (const float*)d_in[0];
    const float* Wq   = (const float*)d_in[1];
    const float* Wdkv = (const float*)d_in[2];
    const float* Wuk  = (const float*)d_in[3];
    const float* Wuv  = (const float*)d_in[4];
    const float* Wo   = (const float*)d_in[5];
    float* out = (float*)d_out;

    char* p = (char*)d_ws;
    auto alloc = [&](size_t bytes) {
        char* r = p; p += (bytes + 255) & ~(size_t)255; return r;
    };
    unsigned short* xb    = (unsigned short*)alloc(8388608ull * 2);  // x bf16 (4096x2048)
    unsigned short* Wqb   = (unsigned short*)alloc(4194304ull * 2);
    unsigned short* Wdkvb = (unsigned short*)alloc(2097152ull * 2);
    unsigned short* Wukb  = (unsigned short*)alloc(1048576ull * 2);
    unsigned short* Wuvb  = (unsigned short*)alloc(1048576ull * 2);
    unsigned short* Wob   = (unsigned short*)alloc(4194304ull * 2);
    float*          dkvf  = (float*)alloc(4194304ull * 4);           // 4096x1024 f32
    unsigned short* dkvb  = (unsigned short*)alloc(4194304ull * 2);
    float*          qfb   = (float*)alloc(8388608ull * 4);           // 4096x2048 f32
    float*          kfb   = (float*)alloc(8388608ull * 4);
    float*          vfb   = (float*)alloc(8388608ull * 4);
    unsigned short* qb    = (unsigned short*)alloc(8388608ull * 2);
    unsigned short* kb    = (unsigned short*)alloc(8388608ull * 2);
    unsigned short* vb    = (unsigned short*)alloc(8388608ull * 2);
    unsigned short* ctx   = (unsigned short*)alloc(8388608ull * 2);

    auto cvt = [&](const float* s, unsigned short* d, int n) {
        cvt_f32_bf16<<<(n + 255) / 256, 256, 0, stream>>>(s, d, n);
    };
    cvt(x,    xb,    8388608);
    cvt(Wq,   Wqb,   4194304);
    cvt(Wdkv, Wdkvb, 2097152);
    cvt(Wuk,  Wukb,  1048576);
    cvt(Wuv,  Wuvb,  1048576);
    cvt(Wo,   Wob,   4194304);

    // q = x @ Wq                (4096x2048x2048)
    gemm_bf16<<<dim3(16, 32), 256, 0, stream>>>(xb, Wqb, qfb, 4096, 2048, 2048, 2048, 2048, 2048);
    // dkv = x @ Wdkv            (4096x1024x2048)
    gemm_bf16<<<dim3(8, 32), 256, 0, stream>>>(xb, Wdkvb, dkvf, 4096, 1024, 2048, 2048, 1024, 1024);
    cvt(dkvf, dkvb, 4194304);
    // k = dk @ Wuk              (A = dkv[:, :512], lda=1024)
    gemm_bf16<<<dim3(16, 32), 256, 0, stream>>>(dkvb, Wukb, kfb, 4096, 2048, 512, 1024, 2048, 2048);
    // v = dv @ Wuv              (A = dkv[:, 512:], lda=1024)
    gemm_bf16<<<dim3(16, 32), 256, 0, stream>>>(dkvb + 512, Wuvb, vfb, 4096, 2048, 512, 1024, 2048, 2048);

    // RoPE(q), RoPE(k), convert v   (4194304 pairs each for rope)
    rope_to_bf16<<<16384, 256, 0, stream>>>(qfb, qb, 4194304);
    rope_to_bf16<<<16384, 256, 0, stream>>>(kfb, kb, 4194304);
    cvt(vfb, vb, 8388608);

    // Fused attention: 16 q-tiles x 32 (b,h) workgroups
    mla_flash<<<dim3(16, 32), 256, 0, stream>>>(qb, kb, vb, ctx);

    // out = ctx @ Wo  -> fp32 straight into d_out
    gemm_bf16<<<dim3(16, 32), 256, 0, stream>>>(ctx, Wob, out, 4096, 2048, 2048, 2048, 2048, 2048);
}
